// Attention_56822417326562
// MI455X (gfx1250) — compile-verified, matrix operand order checked by reading, not run on hardware
//
#include <hip/hip_runtime.h>
#include <hip/hip_bf16.h>

typedef _Float16 half8  __attribute__((ext_vector_type(8)));
typedef _Float16 half16 __attribute__((ext_vector_type(16)));
typedef float    float8 __attribute__((ext_vector_type(8)));

#define BATCH 8
#define DIM 512
#define NTOK 1600
#define HQKV 1024
#define NHEADS 8
#define HH 40
#define WW 40
#define ATT_SCALE 0.17677669529663687f  // 1/sqrt(32)

// ---------------- prep kernels ----------------

__global__ void convert_x_kernel(const float* __restrict__ x, _Float16* __restrict__ xh) {
    int idx = blockIdx.x * blockDim.x + threadIdx.x;
    const int total = BATCH * NTOK * DIM;
    if (idx >= total) return;
    int c = idx & (DIM - 1);
    int n = (idx >> 9) % NTOK;
    int b = idx / (NTOK * DIM);
    xh[idx] = (_Float16)x[((size_t)b * DIM + c) * NTOK + n];
}

__global__ void fold_w_kernel(const float* __restrict__ w, const float* __restrict__ g,
                              const float* __restrict__ be, const float* __restrict__ mu,
                              const float* __restrict__ var, _Float16* __restrict__ wh,
                              float* __restrict__ bias, int O, int K) {
    int idx = blockIdx.x * blockDim.x + threadIdx.x;
    if (idx >= O * K) return;
    int o = idx / K;
    float inv = g[o] * rsqrtf(var[o] + 1e-5f);
    wh[idx] = (_Float16)(w[idx] * inv);
    if ((idx % K) == 0) bias[o] = be[o] - mu[o] * inv;
}

// ---------------- tiled WMMA GEMM ----------------
// act: [B][NTOK][K] f16 (K contiguous), wt: [O][K] f16, out[n,o] = sum_k act*wt + bias[o]
template <bool OUT_F32T>
__global__ void __launch_bounds__(256)
gemm_wmma_kernel(const _Float16* __restrict__ act, const _Float16* __restrict__ wt,
                 const float* __restrict__ bias, _Float16* __restrict__ outh,
                 float* __restrict__ outf, int K, int O) {
    const int nb = blockIdx.x;            // 64-row n tile
    const int ob = blockIdx.y;            // 128-col o tile
    const int b  = blockIdx.z;
    __shared__ __attribute__((aligned(16))) _Float16 lA[64 * 32];
    __shared__ __attribute__((aligned(16))) _Float16 lB[128 * 32];
    const int tid  = threadIdx.x;
    const int lane = tid & 31;
    const int wave = tid >> 5;            // 8 waves
    const int nsub = (wave & 3) * 16;     // wave row base in 64
    const int osub = (wave >> 2) * 64;    // wave col base in 128
    const int l16  = lane & 15;
    const int kb   = (lane >> 4) * 8;
    const int rhi  = (lane >> 4) * 8;

    float8 acc[4] = {};
    const _Float16* actB = act + ((size_t)b * NTOK + (size_t)nb * 64) * K;
    const _Float16* wtB  = wt + (size_t)(ob * 128) * K;

    for (int k0 = 0; k0 < K; k0 += 32) {
        __syncthreads();
        {   // stage A: 64x32 halves, one b128 per thread
            int row = tid >> 2, kc = (tid & 3) * 8;
            *(half8*)&lA[row * 32 + kc] = *(const half8*)&actB[(size_t)row * K + k0 + kc];
            if (k0 + 32 < K)
                __builtin_prefetch(&actB[(size_t)row * K + k0 + 32 + kc], 0, 0);
        }
        // stage B: 128x32 halves, two b128 per thread
        for (int c = tid; c < 512; c += 256) {
            int o = c >> 2, kc = (c & 3) * 8;
            *(half8*)&lB[o * 32 + kc] = *(const half8*)&wtB[(size_t)o * K + k0 + kc];
        }
        __syncthreads();
        half16 af;
        {
            half8 lo = *(const half8*)&lA[(nsub + l16) * 32 + kb];
            half8 hi = *(const half8*)&lA[(nsub + l16) * 32 + kb + 16];
            for (int j = 0; j < 8; ++j) { af[j] = lo[j]; af[8 + j] = hi[j]; }
        }
        for (int t = 0; t < 4; ++t) {
            half16 bf;
            half8 lo = *(const half8*)&lB[(osub + t * 16 + l16) * 32 + kb];
            half8 hi = *(const half8*)&lB[(osub + t * 16 + l16) * 32 + kb + 16];
            for (int j = 0; j < 8; ++j) { bf[j] = lo[j]; bf[8 + j] = hi[j]; }
            acc[t] = __builtin_amdgcn_wmma_f32_16x16x32_f16(false, af, false, bf,
                                                            (short)0, acc[t], false, false);
        }
    }
    for (int t = 0; t < 4; ++t) {
        int o = ob * 128 + osub + t * 16 + l16;
        float bv = bias[o];
        for (int r = 0; r < 8; ++r) {
            int n = nb * 64 + nsub + rhi + r;
            float v = acc[t][r] + bv;
            if (OUT_F32T) outf[((size_t)b * O + o) * NTOK + n] = v;
            else          outh[((size_t)b * NTOK + n) * O + o] = (_Float16)v;
        }
    }
}

// ---------------- flash attention (transposed-S formulation) ----------------
// qkvh: [b][n][1024]; per head: c' 0..31 = q, 32..63 = k, 64..127 = v
// Computes S^T = K x Q^T so softmax over keys is (mostly) in-lane, and the
// S^T accumulator fragment IS the B-fragment for O^T = V x P^T (no LDS round trip).
__global__ void __launch_bounds__(128)
flash_attn_kernel(const _Float16* __restrict__ qkvh, _Float16* __restrict__ attnh) {
    const int nb = blockIdx.x;            // 64 query rows per block
    const int h  = blockIdx.y;
    const int b  = blockIdx.z;
    const int tid  = threadIdx.x;         // 128 threads, 4 waves
    const int lane = tid & 31;
    const int wave = tid >> 5;
    const int l16  = lane & 15;
    const int kb   = (lane >> 4) * 8;
    const int rhi  = (lane >> 4) * 8;
    __shared__ __attribute__((aligned(16))) _Float16 lK[32 * 32];       // [m][kk]
    __shared__ __attribute__((aligned(16))) _Float16 lVT[64 * 32];      // [d][m]

    const size_t rowb = (size_t)b * NTOK;
    const int cq = h * 128, ck = h * 128 + 32, cv = h * 128 + 64;

    // Q as B-fragment: lane owns query column n, halves are kk values
    const int nq = nb * 64 + wave * 16 + l16;
    half16 qf;
    {
        const _Float16* qp = &qkvh[(rowb + nq) * HQKV + cq];
        half8 lo = *(const half8*)&qp[kb];
        half8 hi = *(const half8*)&qp[kb + 16];
        for (int j = 0; j < 8; ++j) { qf[j] = lo[j]; qf[8 + j] = hi[j]; }
    }

    float8 Oacc[4] = {};                  // O^T: M = d (4 tiles of 16), N = query (lane)
    float mi = -1e30f, li = 0.f;          // per-lane (per-query) softmax state

    for (int m0 = 0; m0 < NTOK; m0 += 32) {
        __syncthreads();
        {   // stage K tile [32 m][32 kk], one b128 per thread
            int row = tid >> 2, kc = (tid & 3) * 8;
            *(half8*)&lK[row * 32 + kc] =
                *(const half8*)&qkvh[(rowb + m0 + row) * HQKV + ck + kc];
        }
        {   // stage V transposed: lVT[d][m], packed pair-of-m b32 stores
            int m2 = (tid & 15) * 2, dbase = (tid >> 4) * 8;
            const _Float16* v0 = &qkvh[(rowb + m0 + m2) * HQKV + cv + dbase];
            const _Float16* v1 = &qkvh[(rowb + m0 + m2 + 1) * HQKV + cv + dbase];
            half8 a = *(const half8*)v0;
            half8 c = *(const half8*)v1;
            for (int j = 0; j < 8; ++j) {
                union { _Float16 hh[2]; unsigned u; } pk;
                pk.hh[0] = a[j]; pk.hh[1] = c[j];
                *(unsigned*)&lVT[(dbase + j) * 32 + m2] = pk.u;
            }
        }
        __syncthreads();

        // S^T tiles: St[t] = K(m-tile t, 16x32) x Q^T(32x16)
        float8 St[2];
        for (int t = 0; t < 2; ++t) {
            half16 kf;
            half8 lo = *(const half8*)&lK[(t * 16 + l16) * 32 + kb];
            half8 hi = *(const half8*)&lK[(t * 16 + l16) * 32 + kb + 16];
            for (int j = 0; j < 8; ++j) { kf[j] = lo[j]; kf[8 + j] = hi[j]; }
            float8 z = {};
            St[t] = __builtin_amdgcn_wmma_f32_16x16x32_f16(false, kf, false, qf,
                                                           (short)0, z, false, false);
        }

        // online softmax over keys: in-lane over 16 values + one cross-half shuffle
        float mx = -1e30f;
        for (int t = 0; t < 2; ++t)
            for (int r = 0; r < 8; ++r) mx = fmaxf(mx, St[t][r] * ATT_SCALE);
        mx = fmaxf(mx, __shfl_xor(mx, 16, 32));
        float mnew = fmaxf(mi, mx);
        float corr = __expf(mi - mnew);

        // P^T in-register: S^T C-fragment == B-fragment layout for V x P^T
        half16 pf;
        float ps = 0.f;
        for (int t = 0; t < 2; ++t)
            for (int r = 0; r < 8; ++r) {
                float p = __expf(St[t][r] * ATT_SCALE - mnew);
                ps += p;
                pf[t * 8 + r] = (_Float16)p;
            }
        ps += __shfl_xor(ps, 16, 32);
        li = li * corr + ps;
        mi = mnew;
        for (int t = 0; t < 4; ++t)
            for (int r = 0; r < 8; ++r) Oacc[t][r] *= corr;

        // O^T += V(d-tile,16x32) x P^T(32x16)
        for (int t = 0; t < 4; ++t) {
            half16 vf;
            half8 lo = *(const half8*)&lVT[(t * 16 + l16) * 32 + kb];
            half8 hi = *(const half8*)&lVT[(t * 16 + l16) * 32 + kb + 16];
            for (int j = 0; j < 8; ++j) { vf[j] = lo[j]; vf[8 + j] = hi[j]; }
            Oacc[t] = __builtin_amdgcn_wmma_f32_16x16x32_f16(false, vf, false, pf,
                                                             (short)0, Oacc[t], false, false);
        }
    }

    // epilogue: d runs along VGPRs -> contiguous b128 stores
    float rinv = 1.f / li;
    _Float16* op = &attnh[(rowb + nq) * DIM + h * 64];
    for (int t = 0; t < 4; ++t) {
        half8 oh;
        for (int r = 0; r < 8; ++r) oh[r] = (_Float16)(Oacc[t][r] * rinv);
        *(half8*)&op[t * 16 + rhi] = oh;
    }
}

// ---------------- depthwise 3x3 + BN + add ----------------
__global__ void pe_add_kernel(const _Float16* __restrict__ qkvh, const _Float16* __restrict__ attnh,
                              const float* __restrict__ pw, const float* __restrict__ g,
                              const float* __restrict__ be, const float* __restrict__ mu,
                              const float* __restrict__ var, _Float16* __restrict__ yh) {
    int idx = blockIdx.x * blockDim.x + threadIdx.x;
    const int total = BATCH * NTOK * DIM;
    if (idx >= total) return;
    int c = idx & (DIM - 1);
    int n = (idx >> 9) % NTOK;
    int b = idx / (NTOK * DIM);
    int hh = n / WW, ww = n % WW;
    int hd = c >> 6, d = c & 63;
    int cvo = hd * 128 + 64 + d;
    float acc = 0.f;
    for (int ky = 0; ky < 3; ++ky) {
        int y = hh + ky - 1;
        if (y < 0 || y >= HH) continue;
        for (int kx = 0; kx < 3; ++kx) {
            int xx = ww + kx - 1;
            if (xx < 0 || xx >= WW) continue;
            float vv = (float)qkvh[((size_t)b * NTOK + y * WW + xx) * HQKV + cvo];
            acc += vv * pw[c * 9 + ky * 3 + kx];
        }
    }
    float inv = g[c] * rsqrtf(var[c] + 1e-5f);
    float pe = acc * inv + (be[c] - mu[c] * inv);
    yh[idx] = (_Float16)(pe + (float)attnh[idx]);
}

// ---------------- launcher ----------------
extern "C" void kernel_launch(void* const* d_in, const int* in_sizes, int n_in,
                              void* d_out, int out_size, void* d_ws, size_t ws_size,
                              hipStream_t stream) {
    const float* x          = (const float*)d_in[0];
    const float* qkv_w      = (const float*)d_in[1];
    const float* qkv_gamma  = (const float*)d_in[2];
    const float* qkv_beta   = (const float*)d_in[3];
    const float* qkv_mean   = (const float*)d_in[4];
    const float* qkv_var    = (const float*)d_in[5];
    const float* pe_w       = (const float*)d_in[6];
    const float* pe_gamma   = (const float*)d_in[7];
    const float* pe_beta    = (const float*)d_in[8];
    const float* pe_mean    = (const float*)d_in[9];
    const float* pe_var     = (const float*)d_in[10];
    const float* proj_w     = (const float*)d_in[11];
    const float* proj_gamma = (const float*)d_in[12];
    const float* proj_beta  = (const float*)d_in[13];
    const float* proj_mean  = (const float*)d_in[14];
    const float* proj_var   = (const float*)d_in[15];
    float* out = (float*)d_out;

    char* ws = (char*)d_ws;
    const size_t SZ_XH   = (size_t)BATCH * NTOK * DIM * 2;
    const size_t SZ_QKVH = (size_t)BATCH * NTOK * HQKV * 2;
    const size_t SZ_ATT  = SZ_XH;
    const size_t SZ_YH   = SZ_XH;
    _Float16* xh     = (_Float16*)(ws);
    _Float16* qkvh   = (_Float16*)(ws + SZ_XH);
    _Float16* attnh  = (_Float16*)(ws + SZ_XH + SZ_QKVH);
    _Float16* yh     = (_Float16*)(ws + SZ_XH + SZ_QKVH + SZ_ATT);
    char* wbase      = ws + SZ_XH + SZ_QKVH + SZ_ATT + SZ_YH;
    _Float16* wqkvh  = (_Float16*)(wbase);
    _Float16* wprojh = (_Float16*)(wbase + (size_t)HQKV * DIM * 2);
    float*    bqkv   = (float*)(wbase + (size_t)HQKV * DIM * 2 + (size_t)DIM * DIM * 2);
    float*    bproj  = (float*)((char*)bqkv + HQKV * 4);

    const int total = BATCH * NTOK * DIM;
    convert_x_kernel<<<(total + 255) / 256, 256, 0, stream>>>(x, xh);
    fold_w_kernel<<<(HQKV * DIM + 255) / 256, 256, 0, stream>>>(
        qkv_w, qkv_gamma, qkv_beta, qkv_mean, qkv_var, wqkvh, bqkv, HQKV, DIM);
    fold_w_kernel<<<(DIM * DIM + 255) / 256, 256, 0, stream>>>(
        proj_w, proj_gamma, proj_beta, proj_mean, proj_var, wprojh, bproj, DIM, DIM);

    gemm_wmma_kernel<false><<<dim3(NTOK / 64, HQKV / 128, BATCH), 256, 0, stream>>>(
        xh, wqkvh, bqkv, qkvh, nullptr, DIM, HQKV);

    flash_attn_kernel<<<dim3(NTOK / 64, NHEADS, BATCH), 128, 0, stream>>>(qkvh, attnh);

    pe_add_kernel<<<(total + 255) / 256, 256, 0, stream>>>(
        qkvh, attnh, pe_w, pe_gamma, pe_beta, pe_mean, pe_var, yh);

    gemm_wmma_kernel<true><<<dim3(NTOK / 64, DIM / 128, BATCH), 256, 0, stream>>>(
        yh, wprojh, bproj, nullptr, out, DIM, DIM);

    (void)in_sizes; (void)n_in; (void)out_size; (void)ws_size;
}
